// TemporalBlock_49606872268963
// MI455X (gfx1250) — matrix operand order repaired and checked by reading, not compiled, not run on hardware
//
#include <hip/hip_runtime.h>
#include <hip/hip_bf16.h>

typedef __bf16 bf16;
typedef __attribute__((ext_vector_type(16))) __bf16 v16bf;
typedef __attribute__((ext_vector_type(8)))  __bf16 v8bf;
typedef __attribute__((ext_vector_type(8)))  float  v8f;

#define DEVINL static __device__ __forceinline__

constexpr int Bn   = 128;
constexpr int CIN  = 256;
constexpr int COUT = 512;
constexpr int T    = 200;
constexpr int Tp   = 208;            // T padded to multiple of 16 (13 tiles)
constexpr int TpS  = 224;            // attbf row stride: covers k-block over-read
constexpr int Hh   = 8;
constexpr int Kd   = 64;
constexpr int NQK  = Hh * Kd;        // 512
constexpr int NV   = Hh * CIN;       // 2048
constexpr int NQKV = NQK + NQK + NV; // 3072

DEVINL bf16 tobf(float f) { return (bf16)f; }

// A-fragment: row-major MxK bf16 source. Lane l holds row (l&15).
// element e -> k = (e>>3)*16 + khalf*8 + (e&7)   (ISA 16-bit A 16x32 layout)
DEVINL v16bf ldfragA(const bf16* base, int row, size_t stride, int k0, int khalf) {
  const bf16* p = base + (size_t)row * stride + (size_t)(k0 + khalf * 8);
  v8bf lo = *(const v8bf*)p;
  v8bf hi = *(const v8bf*)(p + 16);
  v16bf f;
#pragma unroll
  for (int e = 0; e < 8; ++e) { f[e] = lo[e]; f[e + 8] = hi[e]; }
  return f;
}

// B-fragment loaded from row-major NxK (i.e. B^T) bf16 source.
// Lane l holds column (l&15); element e -> k = (e>>3)*16 + khalf*8 + (e&7) (matches A)
DEVINL v16bf ldfragB(const bf16* base, int row, size_t stride, int k0, int khalf) {
  const bf16* p = base + (size_t)row * stride + (size_t)(k0 + khalf * 8);
  v8bf lo = *(const v8bf*)p;
  v8bf hi = *(const v8bf*)(p + 16);
  v16bf f;
#pragma unroll
  for (int e = 0; e < 8; ++e) { f[e] = lo[e]; f[e + 8] = hi[e]; }
  return f;
}

DEVINL v8f wmma_bf16(v16bf a, v16bf b, v8f c) {
  return __builtin_amdgcn_wmma_f32_16x16x32_bf16(false, a, false, b, (short)0, c,
                                                 false, false);
}

// ---------------- pack x (B,C,T) f32 -> xp (B,Tp,C) bf16 (zero-padded) -------
__global__ void k_pack_x(const float* __restrict__ x, bf16* __restrict__ xp) {
  int idx = blockIdx.x * blockDim.x + threadIdx.x;
  int c = idx % CIN;
  int t = (idx / CIN) % Tp;
  int b = idx / (CIN * Tp);
  float v = (t < T) ? x[((size_t)b * CIN + c) * T + t] : 0.f;
  xp[idx] = tobf(v);
}

// ---------------- weight prep: transpose+convert to bf16 ---------------------
__global__ void k_prep_w(const float* __restrict__ Wq, const float* __restrict__ Wk,
                         const float* __restrict__ Wv, const float* __restrict__ Wp,
                         const float* __restrict__ Wc, const float* __restrict__ Wd,
                         bf16* __restrict__ WqkvT, bf16* __restrict__ WpT,
                         bf16* __restrict__ WcT, bf16* __restrict__ WdT) {
  int idx = blockIdx.x * blockDim.x + threadIdx.x;
  const int R1 = NQKV * CIN;      // WqkvT[n][k]
  const int R2 = CIN * NV;        // WpT[n][k]
  const int R3 = 3 * COUT * CIN;  // WcT[ks][o][i]
  const int R4 = COUT * CIN;      // WdT[o][i]
  if (idx < R1) {
    int n = idx / CIN, k = idx % CIN;
    float v;
    if (n < NQK)          v = Wq[(size_t)k * NQK + n];
    else if (n < 2 * NQK) v = Wk[(size_t)k * NQK + (n - NQK)];
    else                  v = Wv[(size_t)k * NV + (n - 2 * NQK)];
    WqkvT[idx] = tobf(v);
  } else if (idx < R1 + R2) {
    int j = idx - R1;
    int n = j / NV, k = j % NV;
    WpT[j] = tobf(Wp[(size_t)k * CIN + n]);
  } else if (idx < R1 + R2 + R3) {
    int j = idx - R1 - R2;
    int ks = j / (COUT * CIN);
    int r = j % (COUT * CIN);
    int o = r / CIN, i = r % CIN;
    WcT[j] = tobf(Wc[(size_t)o * (CIN * 3) + i * 3 + ks]);
  } else if (idx < R1 + R2 + R3 + R4) {
    int j = idx - R1 - R2 - R3;
    WdT[j] = tobf(Wd[j]);
  }
}

// ---------------- fused QKV projection GEMM, 32x32 per wave ------------------
// xp (B*Tp, 256) @ WqkvT^T (256, 3072) -> scatter into Q/K (b,h,Tp,64), VT (b,h,256,Tp)
DEVINL void qkv_store(v8f acc, int m0, int nl, const float* bq, const float* bk,
                      const float* bv, bf16* Q, bf16* Kb, bf16* VT, int khalf) {
  float bias;
  if (nl < NQK)          bias = bq[nl];
  else if (nl < 2 * NQK) bias = bk[nl - NQK];
  else                   bias = bv[nl - 2 * NQK];
#pragma unroll
  for (int r = 0; r < 8; ++r) {
    int m = m0 + r + khalf * 8;
    int b_ = m / Tp, t = m - b_ * Tp;
    float v = acc[r] + bias;
    if (nl < NQK) {
      int h = nl >> 6, kk = nl & 63;
      Q[(((size_t)b_ * Hh + h) * Tp + t) * Kd + kk] = tobf(v);
    } else if (nl < 2 * NQK) {
      int n2 = nl - NQK, h = n2 >> 6, kk = n2 & 63;
      Kb[(((size_t)b_ * Hh + h) * Tp + t) * Kd + kk] = tobf(v);
    } else {
      int n2 = nl - 2 * NQK, h = n2 >> 8, cc = n2 & 255;
      VT[(((size_t)b_ * Hh + h) * CIN + cc) * Tp + t] = tobf(v);
    }
  }
}

__global__ void k_qkv(const bf16* __restrict__ xp, const bf16* __restrict__ WqkvT,
                      const float* __restrict__ bq, const float* __restrict__ bk,
                      const float* __restrict__ bv, bf16* __restrict__ Q,
                      bf16* __restrict__ Kb, bf16* __restrict__ VT) {
  const int NT2 = NQKV / 32;  // 96
  int wid = blockIdx.x * (blockDim.x >> 5) + (threadIdx.x >> 5);
  int lane = threadIdx.x & 31, khalf = lane >> 4;
  int mt = wid / NT2, nt = wid % NT2;
  int m0 = mt * 32, n0 = nt * 32;
  v8f a00 = {}, a01 = {}, a10 = {}, a11 = {};
#pragma unroll
  for (int k0 = 0; k0 < CIN; k0 += 32) {
    v16bf A0 = ldfragA(xp, m0 + (lane & 15), CIN, k0, khalf);
    v16bf A1 = ldfragA(xp, m0 + 16 + (lane & 15), CIN, k0, khalf);
    v16bf B0 = ldfragB(WqkvT, n0 + (lane & 15), CIN, k0, khalf);
    v16bf B1 = ldfragB(WqkvT, n0 + 16 + (lane & 15), CIN, k0, khalf);
    a00 = wmma_bf16(A0, B0, a00);
    a01 = wmma_bf16(A0, B1, a01);
    a10 = wmma_bf16(A1, B0, a10);
    a11 = wmma_bf16(A1, B1, a11);
  }
  int nl0 = n0 + (lane & 15), nl1 = n0 + 16 + (lane & 15);
  qkv_store(a00, m0,      nl0, bq, bk, bv, Q, Kb, VT, khalf);
  qkv_store(a01, m0,      nl1, bq, bk, bv, Q, Kb, VT, khalf);
  qkv_store(a10, m0 + 16, nl0, bq, bk, bv, Q, Kb, VT, khalf);
  qkv_store(a11, m0 + 16, nl1, bq, bk, bv, Q, Kb, VT, khalf);
}

// ---------------- scores = Q @ K^T / 8, written into att output --------------
__global__ void k_scores(const bf16* __restrict__ Q, const bf16* __restrict__ Kb,
                         float* __restrict__ att) {
  const int TT = Tp / 16;  // 13
  int wid = blockIdx.x * (blockDim.x >> 5) + (threadIdx.x >> 5);
  int lane = threadIdx.x & 31, khalf = lane >> 4;
  int bh = wid / (TT * TT);
  int rem = wid % (TT * TT);
  int mt = rem / TT, nt = rem % TT;
  if (nt > mt) return;  // tile fully above causal diagonal
  int m0 = mt * 16, n0 = nt * 16;
  const bf16* Qb = Q + (size_t)bh * Tp * Kd;
  const bf16* Kbb = Kb + (size_t)bh * Tp * Kd;
  v8f acc = {};
#pragma unroll
  for (int k0 = 0; k0 < Kd; k0 += 32) {
    v16bf a = ldfragA(Qb, m0 + (lane & 15), Kd, k0, khalf);
    v16bf b = ldfragB(Kbb, n0 + (lane & 15), Kd, k0, khalf);
    acc = wmma_bf16(a, b, acc);
  }
  int n = n0 + (lane & 15);
#pragma unroll
  for (int r = 0; r < 8; ++r) {
    int m = m0 + r + khalf * 8;
    if (m < T && n < T) att[((size_t)bh * T + m) * T + n] = acc[r] * 0.125f;
  }
}

// ------- causal softmax in place + emit zero-padded bf16 copy (stride TpS) ---
__global__ void k_softmax(float* __restrict__ att, bf16* __restrict__ attbf) {
  int lane = threadIdx.x & 31;
  int row = blockIdx.x * (blockDim.x >> 5) + (threadIdx.x >> 5);
  int t = row % Tp;
  int bh = row / Tp;
  bf16* brow = attbf + ((size_t)bh * Tp + t) * TpS;
  if (t >= T) {  // padding row: zero the bf16 copy
    for (int s = lane; s < TpS; s += 32) brow[s] = tobf(0.f);
    return;
  }
  float* rp = att + ((size_t)bh * T + t) * T;
  float mx = -3.0e38f;
  for (int s = lane; s <= t; s += 32) mx = fmaxf(mx, rp[s]);
#pragma unroll
  for (int o = 16; o > 0; o >>= 1) mx = fmaxf(mx, __shfl_xor(mx, o, 32));
  float sum = 0.f;
  for (int s = lane; s <= t; s += 32) {
    float e = __expf(rp[s] - mx);
    rp[s] = e;
    sum += e;
  }
#pragma unroll
  for (int o = 16; o > 0; o >>= 1) sum += __shfl_xor(sum, o, 32);
  float inv = 1.f / sum;
  for (int s = lane; s < TpS; s += 32) {
    float v = (s <= t) ? rp[s] * inv : 0.f;
    if (s < T) rp[s] = v;
    brow[s] = tobf(v);
  }
}

// ---------------- y = att @ V, 16x64 per wave -> Y (b,h,Tp,256) bf16 ---------
__global__ void k_attv(const bf16* __restrict__ attbf, const bf16* __restrict__ VT,
                       bf16* __restrict__ Y) {
  const int NT4 = CIN / 64;  // 4
  const int MT = Tp / 16;    // 13
  int wid = blockIdx.x * (blockDim.x >> 5) + (threadIdx.x >> 5);
  int lane = threadIdx.x & 31, khalf = lane >> 4;
  int bh = wid / (MT * NT4);
  int rem = wid % (MT * NT4);
  int mt = rem / NT4, nt = rem % NT4;
  int m0 = mt * 16, n0 = nt * 64;
  const bf16* Ab = attbf + (size_t)bh * Tp * TpS;
  const bf16* Vb = VT + (size_t)bh * CIN * Tp;
  int kblocks = (m0 + 16 + 31) / 32;  // causal: only s <= m0+15 can be nonzero
  v8f acc0 = {}, acc1 = {}, acc2 = {}, acc3 = {};
  for (int kb = 0; kb < kblocks; ++kb) {
    int k0 = kb * 32;
    v16bf a = ldfragA(Ab, m0 + (lane & 15), TpS, k0, khalf);  // padded zeros beyond T
    v16bf b0 = ldfragB(Vb, n0 + (lane & 15), Tp, k0, khalf);
    v16bf b1 = ldfragB(Vb, n0 + 16 + (lane & 15), Tp, k0, khalf);
    v16bf b2 = ldfragB(Vb, n0 + 32 + (lane & 15), Tp, k0, khalf);
    v16bf b3 = ldfragB(Vb, n0 + 48 + (lane & 15), Tp, k0, khalf);
    acc0 = wmma_bf16(a, b0, acc0);
    acc1 = wmma_bf16(a, b1, acc1);
    acc2 = wmma_bf16(a, b2, acc2);
    acc3 = wmma_bf16(a, b3, acc3);
  }
#pragma unroll
  for (int r = 0; r < 8; ++r) {
    int m = m0 + r + khalf * 8;
    bf16* yr = Y + ((size_t)bh * Tp + m) * CIN + (lane & 15);
    yr[n0]      = tobf(acc0[r]);
    yr[n0 + 16] = tobf(acc1[r]);
    yr[n0 + 32] = tobf(acc2[r]);
    yr[n0 + 48] = tobf(acc3[r]);
  }
}

// ------- attn_out = Y(B,T,2048) @ Wp + bp, 32x32 per wave -> AOut bf16 -------
__global__ void k_proj(const bf16* __restrict__ Y, const bf16* __restrict__ WpT,
                       const float* __restrict__ bp, bf16* __restrict__ AOut) {
  const int NT2 = CIN / 32;  // 8
  int wid = blockIdx.x * (blockDim.x >> 5) + (threadIdx.x >> 5);
  int lane = threadIdx.x & 31, khalf = lane >> 4;
  int mt = wid / NT2, nt = wid % NT2;
  int m0 = mt * 32, n0 = nt * 32;
  // per 16-row subfragment: batch is uniform (Tp % 16 == 0)
  int mA0 = m0 + (lane & 15), mA1 = m0 + 16 + (lane & 15);
  int b0 = mA0 / Tp, tA0 = mA0 - b0 * Tp;
  int b1 = mA1 / Tp, tA1 = mA1 - b1 * Tp;
  const bf16* Yb0 = Y + (size_t)b0 * Hh * Tp * CIN;
  const bf16* Yb1 = Y + (size_t)b1 * Hh * Tp * CIN;
  v8f a00 = {}, a01 = {}, a10 = {}, a11 = {};
  for (int k0 = 0; k0 < NV; k0 += 32) {
    v16bf A0, A1;
#pragma unroll
    for (int g = 0; g < 2; ++g) {
      int kg = k0 + g * 16 + khalf * 8;  // global k -> (h, c)
      int hh = kg >> 8, cc = kg & 255;
      v8bf v0 = *(const v8bf*)(Yb0 + ((size_t)hh * Tp + tA0) * CIN + cc);
      v8bf v1 = *(const v8bf*)(Yb1 + ((size_t)hh * Tp + tA1) * CIN + cc);
#pragma unroll
      for (int e = 0; e < 8; ++e) { A0[g * 8 + e] = v0[e]; A1[g * 8 + e] = v1[e]; }
    }
    v16bf B0 = ldfragB(WpT, n0 + (lane & 15), NV, k0, khalf);
    v16bf B1 = ldfragB(WpT, n0 + 16 + (lane & 15), NV, k0, khalf);
    a00 = wmma_bf16(A0, B0, a00);
    a01 = wmma_bf16(A0, B1, a01);
    a10 = wmma_bf16(A1, B0, a10);
    a11 = wmma_bf16(A1, B1, a11);
  }
  int n0l = n0 + (lane & 15), n1l = n0 + 16 + (lane & 15);
  float bias0 = bp[n0l], bias1 = bp[n1l];
#pragma unroll
  for (int r = 0; r < 8; ++r) {
    int t0r = m0 + r + khalf * 8 - b0 * Tp;
    int t1r = m0 + 16 + r + khalf * 8 - b1 * Tp;
    AOut[((size_t)b0 * Tp + t0r) * CIN + n0l] = tobf(a00[r] + bias0);
    AOut[((size_t)b0 * Tp + t0r) * CIN + n1l] = tobf(a01[r] + bias1);
    AOut[((size_t)b1 * Tp + t1r) * CIN + n0l] = tobf(a10[r] + bias0);
    AOut[((size_t)b1 * Tp + t1r) * CIN + n1l] = tobf(a11[r] + bias1);
  }
}

// ------- dilated causal conv + residual + double ReLU, 32x16 per wave --------
__global__ void k_convres(const bf16* __restrict__ AOut, const bf16* __restrict__ xp,
                          const bf16* __restrict__ WcT, const bf16* __restrict__ WdT,
                          const float* __restrict__ bc, const float* __restrict__ bd,
                          float* __restrict__ out) {
  const int OT2 = COUT / 32;  // 16
  const int TT = Tp / 16;     // 13
  int wid = blockIdx.x * (blockDim.x >> 5) + (threadIdx.x >> 5);
  int lane = threadIdx.x & 31, khalf = lane >> 4;
  int b = wid / (OT2 * TT);
  int rem = wid % (OT2 * TT);
  int ot = rem / TT, tt = rem % TT;
  int o0 = ot * 32, t0 = tt * 16;
  const bf16* Ab = AOut + (size_t)b * Tp * CIN;
  const bf16* Xb = xp + (size_t)b * Tp * CIN;
  v16bf zero;
#pragma unroll
  for (int e = 0; e < 16; ++e) zero[e] = tobf(0.f);

  v8f conv0 = {}, conv1 = {};
#pragma unroll
  for (int ks = 0; ks < 3; ++ks) {
    int tcol = t0 + (lane & 15) - 4 + 2 * ks;  // causal: t' = t - PAD + ks*DIL <= t
    bool ok = tcol >= 0;
    const bf16* Wk_ = WcT + (size_t)ks * COUT * CIN;
#pragma unroll
    for (int k0 = 0; k0 < CIN; k0 += 32) {
      v16bf A0 = ldfragA(Wk_, o0 + (lane & 15), CIN, k0, khalf);
      v16bf A1 = ldfragA(Wk_, o0 + 16 + (lane & 15), CIN, k0, khalf);
      v16bf B = ok ? ldfragB(Ab, tcol, CIN, k0, khalf) : zero;
      conv0 = wmma_bf16(A0, B, conv0);
      conv1 = wmma_bf16(A1, B, conv1);
    }
  }
  v8f res0 = {}, res1 = {};
  int tc2 = t0 + (lane & 15);
#pragma unroll
  for (int k0 = 0; k0 < CIN; k0 += 32) {
    v16bf A0 = ldfragA(WdT, o0 + (lane & 15), CIN, k0, khalf);
    v16bf A1 = ldfragA(WdT, o0 + 16 + (lane & 15), CIN, k0, khalf);
    v16bf B = ldfragB(Xb, tc2, CIN, k0, khalf);
    res0 = wmma_bf16(A0, B, res0);
    res1 = wmma_bf16(A1, B, res1);
  }
  int t = t0 + (lane & 15);
  if (t < T) {
#pragma unroll
    for (int r = 0; r < 8; ++r) {
      int oA = o0 + r + khalf * 8;
      int oB = oA + 16;
      float cA = fmaxf(conv0[r] + bc[oA], 0.f);
      float cB = fmaxf(conv1[r] + bc[oB], 0.f);
      out[((size_t)b * COUT + oA) * T + t] = fmaxf(cA + res0[r] + bd[oA], 0.f);
      out[((size_t)b * COUT + oB) * T + t] = fmaxf(cB + res1[r] + bd[oB], 0.f);
    }
  }
}

extern "C" void kernel_launch(void* const* d_in, const int* in_sizes, int n_in,
                              void* d_out, int out_size, void* d_ws, size_t ws_size,
                              hipStream_t stream) {
  (void)in_sizes; (void)n_in; (void)out_size; (void)ws_size;
  const float* x  = (const float*)d_in[0];
  const float* Wq = (const float*)d_in[1];
  const float* bq = (const float*)d_in[2];
  const float* Wk = (const float*)d_in[3];
  const float* bk = (const float*)d_in[4];
  const float* Wv = (const float*)d_in[5];
  const float* bv = (const float*)d_in[6];
  const float* Wp = (const float*)d_in[7];
  const float* bp = (const float*)d_in[8];
  const float* Wc = (const float*)d_in[9];
  const float* bc = (const float*)d_in[10];
  const float* Wd = (const float*)d_in[11];
  const float* bd = (const float*)d_in[12];

  float* out = (float*)d_out;
  float* att = out + (size_t)Bn * COUT * T;  // att is second output, flat after out

  bf16* base = (bf16*)d_ws;
  size_t off = 0;
  auto alloc = [&](size_t n) {
    bf16* p = base + off;
    off += (n + 63) & ~(size_t)63;
    return p;
  };
  bf16* xp    = alloc((size_t)Bn * Tp * CIN);
  bf16* WqkvT = alloc((size_t)NQKV * CIN);
  bf16* WpT   = alloc((size_t)CIN * NV);
  bf16* WcT   = alloc((size_t)3 * COUT * CIN);
  bf16* WdT   = alloc((size_t)COUT * CIN);
  bf16* Q     = alloc((size_t)Bn * Hh * Tp * Kd);
  bf16* Kb    = alloc((size_t)Bn * Hh * Tp * Kd);
  bf16* VT    = alloc((size_t)Bn * Hh * CIN * Tp);   // Y follows: tail over-read safe
  bf16* Y     = alloc((size_t)Bn * Hh * Tp * CIN);
  bf16* attbf = alloc((size_t)Bn * Hh * Tp * TpS);   // zero-padded bf16 softmax copy
  bf16* AOut  = alloc((size_t)Bn * Tp * CIN + 64);

  k_pack_x<<<(Bn * Tp * CIN) / 256, 256, 0, stream>>>(x, xp);
  {
    int total = NQKV * CIN + CIN * NV + 3 * COUT * CIN + COUT * CIN;
    k_prep_w<<<(total + 255) / 256, 256, 0, stream>>>(Wq, Wk, Wv, Wp, Wc, Wd,
                                                      WqkvT, WpT, WcT, WdT);
  }
  { int tiles = (Bn * Tp / 32) * (NQKV / 32);   // 79872
    k_qkv<<<tiles / 8, 256, 0, stream>>>(xp, WqkvT, bq, bk, bv, Q, Kb, VT); }
  { int tiles = Bn * Hh * 13 * 13;              // 173056
    k_scores<<<tiles / 8, 256, 0, stream>>>(Q, Kb, att); }
  { int rows = Bn * Hh * Tp;                    // 212992 (incl. padding rows)
    k_softmax<<<rows / 8, 256, 0, stream>>>(att, attbf); }
  { int tiles = Bn * Hh * 13 * (CIN / 64);      // 53248
    k_attv<<<tiles / 8, 256, 0, stream>>>(attbf, VT, Y); }
  { int tiles = (Bn * Tp / 32) * (CIN / 32);    // 6656
    k_proj<<<tiles / 8, 256, 0, stream>>>(Y, WpT, bp, AOut); }
  { int tiles = Bn * (COUT / 32) * 13;          // 26624
    k_convres<<<tiles / 8, 256, 0, stream>>>(AOut, xp, WcT, WdT, bc, bd, out); }
}
